// Faceformer_51101520887971
// MI455X (gfx1250) — matrix operand order repaired and checked by reading, not compiled, not run on hardware
//
#include <hip/hip_runtime.h>
#include <hip/hip_bf16.h>
#include <math.h>

// ---------------- model constants ----------------
#define D_MODEL 1024
#define NHEAD   4
#define HEADDIM 256
#define TT      512
#define SS      512
#define NV3     15069
#define DFF     2048
#define AUD     768
#define PERIOD  25

// ---------------- WMMA types ----------------
typedef __bf16 bf16_t;
typedef bf16_t bf16x16 __attribute__((ext_vector_type(16)));
typedef bf16_t bf16x8  __attribute__((ext_vector_type(8)));
typedef float  f32x8   __attribute__((ext_vector_type(8)));

union AF  { bf16x16 v; bf16x8 h[2]; };
union B8  { bf16x8  v; bf16_t e[8]; };
union CF  { f32x8   v; float  e[8]; };

#define LDSPAD 40   // row stride in bf16 elems -> 80B, multiple of 16B

// ---- stage one 64x32 f32 tile pair into registers (16 elems/operand/thread) ----
template <bool AL>
__device__ __forceinline__ void load_tile(const float* __restrict__ aptr,
                                          const float* __restrict__ wptr,
                                          int k0, int ldCol, int K, int kFull,
                                          float* av, float* wv)
{
    const int kb = k0 + ldCol;
    if (k0 < kFull) {                    // full 32-wide tile (block-uniform)
        if constexpr (AL) {
#pragma unroll
            for (int q = 0; q < 4; ++q) {
                float4 a = *(const float4*)(aptr + kb + 4 * q);
                float4 w = *(const float4*)(wptr + kb + 4 * q);
                av[4*q+0] = a.x; av[4*q+1] = a.y; av[4*q+2] = a.z; av[4*q+3] = a.w;
                wv[4*q+0] = w.x; wv[4*q+1] = w.y; wv[4*q+2] = w.z; wv[4*q+3] = w.w;
            }
        } else {
#pragma unroll
            for (int j = 0; j < 16; ++j) { av[j] = aptr[kb + j]; wv[j] = wptr[kb + j]; }
        }
    } else {                             // K tail: clamped addr + value select
#pragma unroll
        for (int j = 0; j < 16; ++j) {
            int kj = kb + j;
            int kc = kj < K ? kj : K - 1;
            float a = aptr[kc];
            float w = wptr[kc];
            av[j] = kj < K ? a : 0.0f;
            wv[j] = kj < K ? w : 0.0f;
        }
    }
}

__device__ __forceinline__ void stage_store(bf16_t (*sA)[LDSPAD], bf16_t (*sW)[LDSPAD],
                                            int ldRow, int ldCol,
                                            const float* av, const float* wv)
{
    B8 pa0, pa1, pw0, pw1;
#pragma unroll
    for (int j = 0; j < 8; ++j) {
        pa0.e[j] = (bf16_t)av[j];  pa1.e[j] = (bf16_t)av[j + 8];
        pw0.e[j] = (bf16_t)wv[j];  pw1.e[j] = (bf16_t)wv[j + 8];
    }
    *(bf16x8*)&sA[ldRow][ldCol]     = pa0.v;
    *(bf16x8*)&sA[ldRow][ldCol + 8] = pa1.v;
    *(bf16x8*)&sW[ldRow][ldCol]     = pw0.v;
    *(bf16x8*)&sW[ldRow][ldCol + 8] = pw1.v;
}

// 32x32 per wave: 2 A-frags x 2 B-frags -> 4 WMMAs per K-step
__device__ __forceinline__ void do_mma(const bf16_t (*sA)[LDSPAD], const bf16_t (*sW)[LDSPAD],
                                       int lane, int waveM, int waveN,
                                       CF& c00, CF& c01, CF& c10, CF& c11)
{
    const int hb   = lane >> 4;
    const int mrow = waveM * 32 + (lane & 15);
    const int ncol = waveN * 32 + (lane & 15);
    AF a0, a1, b0, b1;
    a0.h[0] = *(const bf16x8*)&sA[mrow][hb * 8];
    a0.h[1] = *(const bf16x8*)&sA[mrow][16 + hb * 8];
    a1.h[0] = *(const bf16x8*)&sA[mrow + 16][hb * 8];
    a1.h[1] = *(const bf16x8*)&sA[mrow + 16][16 + hb * 8];
    b0.h[0] = *(const bf16x8*)&sW[ncol][hb * 8];
    b0.h[1] = *(const bf16x8*)&sW[ncol][16 + hb * 8];
    b1.h[0] = *(const bf16x8*)&sW[ncol + 16][hb * 8];
    b1.h[1] = *(const bf16x8*)&sW[ncol + 16][16 + hb * 8];
    c00.v = __builtin_amdgcn_wmma_f32_16x16x32_bf16(false, a0.v, false, b0.v, (short)0, c00.v, false, false);
    c01.v = __builtin_amdgcn_wmma_f32_16x16x32_bf16(false, a0.v, false, b1.v, (short)0, c01.v, false, false);
    c10.v = __builtin_amdgcn_wmma_f32_16x16x32_bf16(false, a1.v, false, b0.v, (short)0, c10.v, false, false);
    c11.v = __builtin_amdgcn_wmma_f32_16x16x32_bf16(false, a1.v, false, b1.v, (short)0, c11.v, false, false);
}

__device__ __forceinline__ void store_cf(const CF& c, float* __restrict__ C,
                                         const float* __restrict__ bias,
                                         const float* __restrict__ bias2,
                                         const float* __restrict__ resid,
                                         int M, int N, int rbase, int col, int relu)
{
    if (col >= N) return;
    float badd = (bias ? bias[col] : 0.0f) + (bias2 ? bias2[col] : 0.0f);
#pragma unroll
    for (int r = 0; r < 8; ++r) {
        int row = rbase + r;
        if (row < M) {
            float vv = c.e[r] + badd;
            if (relu)  vv = fmaxf(vv, 0.0f);
            if (resid) vv += resid[(size_t)row * N + col];
            C[(size_t)row * N + col] = vv;
        }
    }
}

// =====================================================================
// C[M,N] = act( A[M,K] @ W[N,K]^T + bias + bias2 ) + resid
// bf16 WMMA, f32 accumulate. 64x64 block tile, 128 threads = 4 waves
// (2x2), each wave owns 32x32 (4 WMMAs/K-step). LDS double-buffered,
// one barrier per K-step.
// =====================================================================
template <bool AL>
__global__ __launch_bounds__(128)
void gemm_bf16_wmma(const float* __restrict__ A, const float* __restrict__ W,
                    const float* __restrict__ bias, const float* __restrict__ bias2,
                    const float* __restrict__ resid, float* __restrict__ C,
                    int M, int N, int K, int relu)
{
    __shared__ __align__(16) bf16_t sA[2][64][LDSPAD];
    __shared__ __align__(16) bf16_t sW[2][64][LDSPAD];

    const int tid    = threadIdx.x;
    const int lane   = tid & 31;
    const int wave   = tid >> 5;          // 0..3
    const int waveM  = wave >> 1;         // 0..1
    const int waveN  = wave & 1;          // 0..1
    const int blockM = blockIdx.y * 64;
    const int blockN = blockIdx.x * 64;

    CF c00, c01, c10, c11;
#pragma unroll
    for (int i = 0; i < 8; ++i) { c00.e[i] = 0.0f; c01.e[i] = 0.0f; c10.e[i] = 0.0f; c11.e[i] = 0.0f; }

    const int ldRow = tid >> 1;           // 0..63
    const int ldCol = (tid & 1) << 4;     // 0,16
    // clamp rows: always-in-bounds addresses; junk rows/cols discarded in epilogue
    const int gm = blockM + ldRow; const int gmc = gm < M ? gm : M - 1;
    const int gn = blockN + ldRow; const int gnc = gn < N ? gn : N - 1;
    const float* aptr = A + (size_t)gmc * K;
    const float* wptr = W + (size_t)gnc * K;

    const int kFull = K & ~31;

    float av[16], wv[16];
    load_tile<AL>(aptr, wptr, 0, ldCol, K, kFull, av, wv);
    stage_store(sA[0], sW[0], ldRow, ldCol, av, wv);
    __syncthreads();

    int cur = 0;
    for (int k0 = 32; k0 < K; k0 += 32) {
        load_tile<AL>(aptr, wptr, k0, ldCol, K, kFull, av, wv);           // prefetch next
        do_mma(sA[cur], sW[cur], lane, waveM, waveN, c00, c01, c10, c11); // compute current
        stage_store(sA[cur ^ 1], sW[cur ^ 1], ldRow, ldCol, av, wv);
        __syncthreads();
        cur ^= 1;
    }
    do_mma(sA[cur], sW[cur], lane, waveM, waveN, c00, c01, c10, c11);

    // ---- epilogue ----
    const int col0  = blockN + waveN * 32 + (lane & 15);
    const int col1  = col0 + 16;
    const int rbase = blockM + waveM * 32 + ((lane >> 4) << 3);
    store_cf(c00, C, bias, bias2, resid, M, N, rbase,      col0, relu);
    store_cf(c01, C, bias, bias2, resid, M, N, rbase,      col1, relu);
    store_cf(c10, C, bias, bias2, resid, M, N, rbase + 16, col0, relu);
    store_cf(c11, C, bias, bias2, resid, M, N, rbase + 16, col1, relu);
}

// =====================================================================
// vin = concat([template, vertice[:,:-1]]) - template   (row 0 == 0)
// =====================================================================
__global__ __launch_bounds__(256)
void build_vin_kernel(const float* __restrict__ vertice, const float* __restrict__ tmpl,
                      float* __restrict__ vin)
{
    size_t idx = (size_t)blockIdx.x * 256 + threadIdx.x;
    if (idx >= (size_t)TT * NV3) return;
    int t = (int)(idx / NV3);
    int j = (int)(idx % NV3);
    vin[idx] = (t == 0) ? 0.0f : (vertice[(size_t)(t - 1) * NV3 + j] - tmpl[j]);
}

// =====================================================================
// x[t,d] += style[d] + PE[t % 25, d]   (style = one_hot @ W_obj^T)
// =====================================================================
__global__ __launch_bounds__(256)
void style_pe_kernel(const float* __restrict__ one_hot, const float* __restrict__ W_obj,
                     float* __restrict__ x)
{
    const int t = blockIdx.y;
    const int d = blockIdx.x * 256 + threadIdx.x;
    float style = 0.0f;
#pragma unroll
    for (int j = 0; j < 8; ++j) style += one_hot[j] * W_obj[d * 8 + j];
    const int   p   = t % PERIOD;
    const float div = __expf(-(float)(d & ~1) * (9.210340371976184f / 1024.0f)); // ln(1e4)/D
    const float arg = (float)p * div;
    const float pe  = (d & 1) ? __cosf(arg) : __sinf(arg);
    x[(size_t)t * D_MODEL + d] += style + pe;
}

// =====================================================================
// Self-attention with causal + ALIBI mask.
// mask(h,t,s) = -0.25^(h+1) * floor((t-s)/25)  for s<=t, else -inf.
// One block per (head, tq); 256 threads.
// =====================================================================
__global__ __launch_bounds__(256)
void attn_sa_kernel(const float* __restrict__ qkv, float* __restrict__ ctx)
{
    const int h  = blockIdx.x;
    const int tq = blockIdx.y;
    __shared__ float sq[HEADDIM];
    __shared__ float sp[SS];
    __shared__ float rbuf[256];
    const int tid = threadIdx.x;

    sq[tid] = qkv[(size_t)tq * 3 * D_MODEL + h * HEADDIM + tid];
    __syncthreads();

    const float scale = 0.0625f;                       // 1/sqrt(256)
    const float slope = exp2f(-2.0f * (float)(h + 1)); // 0.25^(h+1)

    for (int s = tid; s < SS; s += 256) {
        const float* kp = qkv + (size_t)s * 3 * D_MODEL + D_MODEL + h * HEADDIM;
        float acc = 0.0f;
#pragma unroll 4
        for (int d = 0; d < HEADDIM; ++d) acc = fmaf(sq[d], kp[d], acc);
        sp[s] = (s <= tq) ? (acc * scale - slope * (float)((tq - s) / PERIOD))
                          : -3.0e38f;
    }
    __syncthreads();

    rbuf[tid] = fmaxf(sp[tid], sp[tid + 256]);
    __syncthreads();
    for (int o = 128; o > 0; o >>= 1) {
        if (tid < o) rbuf[tid] = fmaxf(rbuf[tid], rbuf[tid + o]);
        __syncthreads();
    }
    const float mx = rbuf[0];
    __syncthreads();

    float e0 = __expf(sp[tid] - mx);
    float e1 = __expf(sp[tid + 256] - mx);
    sp[tid] = e0; sp[tid + 256] = e1;
    rbuf[tid] = e0 + e1;
    __syncthreads();
    for (int o = 128; o > 0; o >>= 1) {
        if (tid < o) rbuf[tid] += rbuf[tid + o];
        __syncthreads();
    }
    const float inv = 1.0f / rbuf[0];
    __syncthreads();

    float acc = 0.0f;
    for (int s = 0; s <= tq; ++s)
        acc = fmaf(sp[s], qkv[(size_t)s * 3 * D_MODEL + 2 * D_MODEL + h * HEADDIM + tid], acc);
    ctx[(size_t)tq * D_MODEL + h * HEADDIM + tid] = acc * inv;
}

// =====================================================================
// LayerNorm over D=1024, one block per row, eps = 1e-5
// =====================================================================
__global__ __launch_bounds__(256)
void layernorm_kernel(const float* __restrict__ x, const float* __restrict__ g,
                      const float* __restrict__ b, float* __restrict__ out)
{
    const int row = blockIdx.x;
    const int tid = threadIdx.x;
    __shared__ float rbuf[256];
    const float* xr = x + (size_t)row * D_MODEL;
    float v0 = xr[tid], v1 = xr[tid + 256], v2 = xr[tid + 512], v3 = xr[tid + 768];

    rbuf[tid] = v0 + v1 + v2 + v3;
    __syncthreads();
    for (int o = 128; o > 0; o >>= 1) { if (tid < o) rbuf[tid] += rbuf[tid + o]; __syncthreads(); }
    const float mean = rbuf[0] * (1.0f / D_MODEL);
    __syncthreads();

    float d0 = v0 - mean, d1 = v1 - mean, d2 = v2 - mean, d3 = v3 - mean;
    rbuf[tid] = d0 * d0 + d1 * d1 + d2 * d2 + d3 * d3;
    __syncthreads();
    for (int o = 128; o > 0; o >>= 1) { if (tid < o) rbuf[tid] += rbuf[tid + o]; __syncthreads(); }
    const float rstd = rsqrtf(rbuf[0] * (1.0f / D_MODEL) + 1e-5f);

    float* outr = out + (size_t)row * D_MODEL;
    outr[tid]       = d0 * rstd * g[tid]       + b[tid];
    outr[tid + 256] = d1 * rstd * g[tid + 256] + b[tid + 256];
    outr[tid + 512] = d2 * rstd * g[tid + 512] + b[tid + 512];
    outr[tid + 768] = d3 * rstd * g[tid + 768] + b[tid + 768];
}

// ---------------- host-side GEMM launcher ----------------
static void launch_gemm(hipStream_t s, const float* A, const float* W,
                        const float* bias, const float* bias2, const float* resid,
                        float* C, int M, int N, int K, int relu)
{
    dim3 grid((N + 63) / 64, (M + 63) / 64), blk(128);
    if ((K & 3) == 0)
        gemm_bf16_wmma<true><<<grid, blk, 0, s>>>(A, W, bias, bias2, resid, C, M, N, K, relu);
    else
        gemm_bf16_wmma<false><<<grid, blk, 0, s>>>(A, W, bias, bias2, resid, C, M, N, K, relu);
}

extern "C" void kernel_launch(void* const* d_in, const int* in_sizes, int n_in,
                              void* d_out, int out_size, void* d_ws, size_t ws_size,
                              hipStream_t stream)
{
    const float* audio_feat = (const float*)d_in[0];
    const float* vertice    = (const float*)d_in[1];
    const float* tmpl       = (const float*)d_in[2];
    const float* one_hot    = (const float*)d_in[3];
    const float* W_af   = (const float*)d_in[4];
    const float* b_af   = (const float*)d_in[5];
    const float* W_vm   = (const float*)d_in[6];
    const float* b_vm   = (const float*)d_in[7];
    const float* W_obj  = (const float*)d_in[8];
    const float* Wqkv_sa = (const float*)d_in[9];
    const float* bqkv_sa = (const float*)d_in[10];
    const float* Wo_sa  = (const float*)d_in[11];
    const float* bo_sa  = (const float*)d_in[12];
    const float* Wqkv_ca = (const float*)d_in[13];
    const float* bqkv_ca = (const float*)d_in[14];
    const float* Wo_ca  = (const float*)d_in[15];
    const float* bo_ca  = (const float*)d_in[16];
    const float* W1     = (const float*)d_in[17];
    const float* b1     = (const float*)d_in[18];
    const float* W2     = (const float*)d_in[19];
    const float* b2     = (const float*)d_in[20];
    const float* g1     = (const float*)d_in[21];
    const float* be1    = (const float*)d_in[22];
    const float* g2     = (const float*)d_in[23];
    const float* be2    = (const float*)d_in[24];
    const float* g3     = (const float*)d_in[25];
    const float* be3    = (const float*)d_in[26];
    const float* W_vr   = (const float*)d_in[27];
    const float* b_vr   = (const float*)d_in[28];
    float* out = (float*)d_out;

    // workspace carving (~52 MB of f32)
    float* ws = (float*)d_ws;
    float* vin_raw = ws; ws += (size_t)TT * NV3;
    float* hidden  = ws; ws += (size_t)TT * D_MODEL;
    float* xbuf    = ws; ws += (size_t)TT * D_MODEL;
    float* qkv     = ws; ws += (size_t)TT * 3 * D_MODEL;
    float* attn    = ws; ws += (size_t)TT * D_MODEL;
    float* tmp     = ws; ws += (size_t)TT * D_MODEL;
    float* ffbuf   = ws; ws += (size_t)TT * DFF;
    float* x2      = ws; ws += (size_t)TT * D_MODEL;

    dim3 blk(256);

    // 1) motion input
    {
        size_t n = (size_t)TT * NV3;
        build_vin_kernel<<<(unsigned)((n + 255) / 256), blk, 0, stream>>>(vertice, tmpl, vin_raw);
    }
    // 2) audio hidden = audio_feat @ W_af^T + b_af
    launch_gemm(stream, audio_feat, W_af, b_af, nullptr, nullptr, hidden, TT, D_MODEL, AUD, 0);
    // 3) x = vin_raw @ W_vm^T + b_vm   (big GEMM, K = 15069, unaligned variant)
    launch_gemm(stream, vin_raw, W_vm, b_vm, nullptr, nullptr, xbuf, TT, D_MODEL, NV3, 0);
    // 4) + style + positional encoding
    style_pe_kernel<<<dim3(D_MODEL / 256, TT), blk, 0, stream>>>(one_hot, W_obj, xbuf);
    // 5) self-attention QKV
    launch_gemm(stream, xbuf, Wqkv_sa, bqkv_sa, nullptr, nullptr, qkv, TT, 3 * D_MODEL, D_MODEL, 0);
    // 6) ALIBI-masked attention
    attn_sa_kernel<<<dim3(NHEAD, TT), blk, 0, stream>>>(qkv, attn);
    // 7) O-projection + residual
    launch_gemm(stream, attn, Wo_sa, bo_sa, nullptr, xbuf, tmp, TT, D_MODEL, D_MODEL, 0);
    // 8) LN1
    layernorm_kernel<<<TT, blk, 0, stream>>>(tmp, g1, be1, xbuf);
    // 9) cross-attention: mem_mask=~eye => attention is identity; only V-proj needed
    launch_gemm(stream, hidden, Wqkv_ca + (size_t)2 * D_MODEL * D_MODEL,
                bqkv_ca + 2 * D_MODEL, nullptr, nullptr, attn, TT, D_MODEL, D_MODEL, 0);
    // 10) O-projection + residual
    launch_gemm(stream, attn, Wo_ca, bo_ca, nullptr, xbuf, tmp, TT, D_MODEL, D_MODEL, 0);
    // 11) LN2
    layernorm_kernel<<<TT, blk, 0, stream>>>(tmp, g2, be2, xbuf);
    // 12) FF1 (ReLU)
    launch_gemm(stream, xbuf, W1, b1, nullptr, nullptr, ffbuf, TT, DFF, D_MODEL, 1);
    // 13) FF2 + residual
    launch_gemm(stream, ffbuf, W2, b2, nullptr, xbuf, tmp, TT, D_MODEL, DFF, 0);
    // 14) LN3
    layernorm_kernel<<<TT, blk, 0, stream>>>(tmp, g3, be3, x2);
    // 15) vertex readout: out = x2 @ W_vr^T + b_vr + template (bias2 broadcast)
    launch_gemm(stream, x2, W_vr, b_vr, tmpl, nullptr, out, TT, NV3, D_MODEL, 0);
}
